// FCOS_60266981098210
// MI455X (gfx1250) — matrix operand order repaired and checked by reading, not compiled, not run on hardware
//
#include <hip/hip_runtime.h>
#include <math.h>

typedef _Float16 v16h __attribute__((ext_vector_type(16)));
typedef float    v8f  __attribute__((ext_vector_type(8)));

#define CH 256
#define NGT 40
#define NCLS 20

// ---------------------------------------------------------------------------
// Utility: zero the 4 loss accumulators
// ---------------------------------------------------------------------------
__global__ void zero4_kernel(float* acc) {
    if (threadIdx.x < 4) acc[threadIdx.x] = 0.0f;
}

// ---------------------------------------------------------------------------
// f32 -> f16 elementwise convert (for FPN lateral outputs entering WMMA chain)
// ---------------------------------------------------------------------------
__global__ void cvt_f16_kernel(const float* __restrict__ x,
                               _Float16* __restrict__ y, int n) {
    int i = blockIdx.x * blockDim.x + threadIdx.x;
    if (i < n) y[i] = (_Float16)x[i];
}

// ---------------------------------------------------------------------------
// Generic direct NCHW conv (k=1 or 3), optional stride/ReLU. Used for the
// backbone (Cin<=160), 1x1 laterals and the tiny prediction heads.
// ---------------------------------------------------------------------------
__global__ void conv_nchw_kernel(const float* __restrict__ x,
                                 const float* __restrict__ w,
                                 const float* __restrict__ bias,
                                 float* __restrict__ y,
                                 int B, int Cin, int Hin, int Win,
                                 int Cout, int Hout, int Wout,
                                 int K, int stride, int relu) {
    int idx = blockIdx.x * blockDim.x + threadIdx.x;
    int total = B * Cout * Hout * Wout;
    if (idx >= total) return;
    int wo = idx % Wout; int t = idx / Wout;
    int ho = t % Hout;  t /= Hout;
    int co = t % Cout;  int b = t / Cout;
    int pad = K / 2;
    float acc = bias[co];
    const float* wp = w + (size_t)co * Cin * K * K;
    for (int ci = 0; ci < Cin; ++ci) {
        const float* xp = x + ((size_t)(b * Cin + ci) * Hin) * Win;
        const float* wc = wp + ci * K * K;
        for (int ky = 0; ky < K; ++ky) {
            int iy = ho * stride + ky - pad;
            if ((unsigned)iy >= (unsigned)Hin) continue;
            for (int kx = 0; kx < K; ++kx) {
                int ix = wo * stride + kx - pad;
                if ((unsigned)ix >= (unsigned)Win) continue;
                acc += xp[iy * Win + ix] * wc[ky * K + kx];
            }
        }
    }
    if (relu) acc = fmaxf(acc, 0.0f);
    y[idx] = acc;
}

// ---------------------------------------------------------------------------
// In-place FPN merge: lat[b,c,h,w] += small[b,c,h/2,w/2]  (2x nearest up)
// ---------------------------------------------------------------------------
__global__ void add_up2_kernel(float* __restrict__ lat,
                               const float* __restrict__ small,
                               int B, int C, int H, int W) {
    int idx = blockIdx.x * blockDim.x + threadIdx.x;
    int total = B * C * H * W;
    if (idx >= total) return;
    int w = idx % W; int t = idx / W;
    int h = t % H;  t /= H;
    int c = t % C;  int b = t / C;
    int Hs = H >> 1, Ws = W >> 1;
    lat[idx] += small[((size_t)(b * C + c) * Hs + (h >> 1)) * Ws + (w >> 1)];
}

// ---------------------------------------------------------------------------
// Pack f32 [256,256,3,3] weights -> f16 B-fragment layout:
//   dst[(coTile)(16)][tap(9)][cinChunk(8)][kGroup(2)][n(16)][k(16)]
// so each wmma lane reads its 16 B halves as two contiguous b128 loads.
// ---------------------------------------------------------------------------
__global__ void pack_w_kernel(const float* __restrict__ w,
                              _Float16* __restrict__ wp) {
    int i = blockIdx.x * blockDim.x + threadIdx.x;   // over 256*256*9
    if (i >= CH * CH * 9) return;
    int kx = i % 3; int t = i / 3;
    int ky = t % 3; t /= 3;
    int ci = t % CH; int co = t / CH;
    int tap = ky * 3 + kx;
    int dst = ((((co >> 4) * 9 + tap) * 8 + (ci >> 5)) * 2 + ((ci >> 4) & 1)) * 256
              + (co & 15) * 16 + (ci & 15);
    wp[dst] = (_Float16)w[i];
}

// ---------------------------------------------------------------------------
// Implicit-GEMM 3x3 conv, Cin=Cout=256, stride 1, pad 1, f16 in / f16(+f32)
// out, via v_wmma_f32_16x16x32_f16.  Block = 128 thr = 4 waves.
// Block computes 16 output pixels (row segment) x 128 output channels:
// each wave owns two 16x16 tiles (one A fragment feeds two WMMAs).
// Per cin-chunk a 3-row x 18-px x 32-ch slab is staged once in LDS and
// serves all 9 taps (zero padding resolved at staging time).
// grid = ((W/16)*2, H, B)
// ---------------------------------------------------------------------------
__global__ void __launch_bounds__(128)
conv3x3_wmma_kernel(const _Float16* __restrict__ x16,
                    const _Float16* __restrict__ wpack,
                    const float* __restrict__ bias,
                    _Float16* __restrict__ y16,
                    float* __restrict__ y32,
                    int H, int W, int relu, int write32) {
    __shared__ _Float16 slab[3 * 18 * 32];   // [row 0..2][x 0..17][k 0..31]

    const int tid  = threadIdx.x;
    const int wave = tid >> 5;
    const int lane = tid & 31;
    const int g    = lane >> 4;        // lane group (0/1)
    const int mn   = lane & 15;        // A: pixel row m ; B/C: column n

    const int nx = W >> 4;
    const int xt = blockIdx.x % nx;
    const int cg = blockIdx.x / nx;    // 128-channel output group (0/1)
    const int ct0 = cg * 8 + wave * 2; // this wave's two 16-cout tiles
    const int yrow = blockIdx.y;
    const int b    = blockIdx.z;
    const int x0   = xt * 16;

    v8f acc0 = {};
    v8f acc1 = {};

    for (int cc = 0; cc < 8; ++cc) {
        __syncthreads();
        // Stage slab: threads 0..95 each copy one (row, channel) line of 18px
        if (tid < 96) {
            const int r = tid / 32;        // 0..2 -> input row yrow-1+r
            const int k = tid & 31;        // local cin
            const int iy = yrow + r - 1;
            const bool rowOK = (unsigned)iy < (unsigned)H;
            const _Float16* src =
                x16 + ((size_t)(b * CH + cc * 32 + k) * H + (rowOK ? iy : 0)) * W;
            for (int xx = 0; xx < 18; ++xx) {
                const int ix = x0 + xx - 1;
                _Float16 v = (_Float16)0.0f;
                if (rowOK && (unsigned)ix < (unsigned)W) v = src[ix];
                slab[(r * 18 + xx) * 32 + k] = v;
            }
        }
        __syncthreads();

        #pragma unroll
        for (int tap = 0; tap < 9; ++tap) {
            const int dy = tap / 3;        // 0..2
            const int dx = tap % 3;        // 0..2
            // A fragment (contiguous k order): lane group 0 -> K {0-15},
            // group 1 -> K {16-31}; pixel row m = mn.
            const _Float16* ap = &slab[(dy * 18 + mn + dx) * 32];
            v16h a;
            #pragma unroll
            for (int h = 0; h < 8; ++h) a[h] = ap[16 * g + h];
            #pragma unroll
            for (int h = 0; h < 8; ++h) a[8 + h] = ap[16 * g + 8 + h];

            // Two B fragments: contiguous halves from packed weights
            const size_t wb0 =
                ((size_t)(((ct0 + 0) * 9 + tap) * 8 + cc) * 2 + g) * 256 + mn * 16;
            const size_t wb1 =
                ((size_t)(((ct0 + 1) * 9 + tap) * 8 + cc) * 2 + g) * 256 + mn * 16;
            v16h b0 = *(const v16h*)(wpack + wb0);
            v16h b1 = *(const v16h*)(wpack + wb1);

            acc0 = __builtin_amdgcn_wmma_f32_16x16x32_f16(
                false, a, false, b0, (short)0, acc0, false, false);
            acc1 = __builtin_amdgcn_wmma_f32_16x16x32_f16(
                false, a, false, b1, (short)0, acc1, false, false);
        }
    }

    // C/D layout: VGPR r, lane l -> pixel m = r + 8*(l/16), channel n = l%16
    #pragma unroll
    for (int t2 = 0; t2 < 2; ++t2) {
        v8f acc = (t2 == 0) ? acc0 : acc1;
        const int co = (ct0 + t2) * 16 + mn;
        const float bv = bias[co];
        const size_t base = ((size_t)(b * CH + co) * H + yrow) * W + x0 + 8 * g;
        #pragma unroll
        for (int r = 0; r < 8; ++r) {
            float v = acc[r] + bv;
            if (relu) v = fmaxf(v, 0.0f);
            y16[base + r] = (_Float16)v;
            if (write32) y32[base + r] = v;
        }
    }
}

// ---------------------------------------------------------------------------
// Per-level matching + losses.  One thread per location.
// acc[0]=loss_cls, acc[1]=loss_box, acc[2]=loss_ctr, acc[3]=fg count
// grid = (ceil(HW/256), B)
// ---------------------------------------------------------------------------
__global__ void __launch_bounds__(256)
fcos_loss_kernel(const float* __restrict__ pcls,
                 const float* __restrict__ pbox,
                 const float* __restrict__ pctr,
                 const float* __restrict__ gt,
                 int H, int W, float stride, float lo, float hi,
                 float* __restrict__ acc) {
    __shared__ float gts[NGT * 5];
    __shared__ float red[4];
    const int tid = threadIdx.x;
    const int b = blockIdx.y;
    if (tid < NGT * 5) gts[tid] = gt[b * NGT * 5 + tid];
    if (tid < 4) red[tid] = 0.0f;
    __syncthreads();

    const int HW = H * W;
    const int loc = blockIdx.x * 256 + tid;
    float lc = 0.0f, lb = 0.0f, lt = 0.0f, fgc = 0.0f;

    if (loc < HW) {
        const int h = loc / W, w = loc % W;
        const float lx = (w + 0.5f) * stride;
        const float ly = (h + 0.5f) * stride;

        // argmax match over GT (first-max tie break like jnp.argmax)
        float bestQ = 0.0f;
        float mx1 = -1.0f, my1 = -1.0f, mx2 = -1.0f, my2 = -1.0f, mcl = -1.0f;
        for (int j = 0; j < NGT; ++j) {
            const float x1 = gts[j * 5 + 0], y1 = gts[j * 5 + 1];
            const float x2 = gts[j * 5 + 2], y2 = gts[j * 5 + 3];
            const float d0 = lx - x1, d1 = ly - y1, d2 = x2 - lx, d3 = y2 - ly;
            const float mind = fminf(fminf(d0, d1), fminf(d2, d3));
            const float maxd = fmaxf(fmaxf(d0, d1), fmaxf(d2, d3));
            const bool ok = (mind > 0.0f) && (maxd > lo) && (maxd < hi);
            const float area = (x2 - x1) * (y2 - y1);
            const float q = ok ? (1.0e8f - area) : 0.0f;
            if (q > bestQ) {
                bestQ = q;
                mx1 = x1; my1 = y1; mx2 = x2; my2 = y2; mcl = gts[j * 5 + 4];
            }
        }
        if (bestQ < 1.0e-5f) { mx1 = my1 = mx2 = my2 = mcl = -1.0f; }

        // deltas (or -1 for background)
        float gd[4];
        if (mx1 == -1.0f) {
            gd[0] = gd[1] = gd[2] = gd[3] = -1.0f;
        } else {
            gd[0] = (lx - mx1) / stride; gd[1] = (ly - my1) / stride;
            gd[2] = (mx2 - lx) / stride; gd[3] = (my2 - ly) / stride;
        }
        const bool fg = (mcl >= 0.0f);
        const int  cl = fg ? (int)mcl : -1;
        if (fg) fgc = 1.0f;

        // focal classification loss over all classes
        for (int c = 0; c < NCLS; ++c) {
            const float xl = pcls[((size_t)(b * NCLS + c) * H + h) * W + w];
            const float t = (fg && c == cl) ? 1.0f : 0.0f;
            const float p = 1.0f / (1.0f + expf(-xl));
            const float ce = fmaxf(xl, 0.0f) - xl * t + log1pf(expf(-fabsf(xl)));
            const float pt = p * t + (1.0f - p) * (1.0f - t);
            const float at = 0.25f * t + 0.75f * (1.0f - t);
            const float om = 1.0f - pt;
            lc += at * ce * om * om;
        }
        // box L1
        for (int c = 0; c < 4; ++c) {
            if (gd[c] >= 0.0f) {
                const float pb = pbox[((size_t)(b * 4 + c) * H + h) * W + w];
                lb += 0.25f * fabsf(pb - gd[c]);
            }
        }
        // centerness BCE
        float ctr_t = -1.0f;
        if (gd[0] >= 0.0f) {
            const float num = fminf(gd[0], gd[2]) * fminf(gd[1], gd[3]);
            const float den = fmaxf(gd[0], gd[2]) * fmaxf(gd[1], gd[3]);
            ctr_t = sqrtf(fmaxf(num / (den + 1.0e-8f), 0.0f));
        }
        if (ctr_t >= 0.0f) {
            const float xl = pctr[((size_t)b * H + h) * W + w];
            lt += fmaxf(xl, 0.0f) - xl * ctr_t + log1pf(expf(-fabsf(xl)));
        }
    }

    atomicAdd(&red[0], lc);
    atomicAdd(&red[1], lb);
    atomicAdd(&red[2], lt);
    atomicAdd(&red[3], fgc);
    __syncthreads();
    if (tid == 0) {
        atomicAdd(&acc[0], red[0]);
        atomicAdd(&acc[1], red[1]);
        atomicAdd(&acc[2], red[2]);
        atomicAdd(&acc[3], red[3]);
    }
}

__global__ void finalize_kernel(const float* __restrict__ acc,
                                float* __restrict__ out) {
    if (threadIdx.x == 0) {
        const float n = fmaxf(acc[3], 1.0f);
        out[0] = acc[0] / n;
        out[1] = acc[1] / n;
        out[2] = acc[2] / n;
    }
}

// ---------------------------------------------------------------------------
// Host orchestration
// ---------------------------------------------------------------------------
extern "C" void kernel_launch(void* const* d_in, const int* in_sizes, int n_in,
                              void* d_out, int out_size, void* d_ws, size_t ws_size,
                              hipStream_t stream) {
    (void)in_sizes; (void)n_in; (void)out_size;
    const int B = 4;

    // ---- inputs (jax tree flatten: dict keys sorted alphabetically) ----
    // backbone, fpn_lat, fpn_out, pred_box, pred_cls, pred_ctr, stem_box, stem_cls
    const float* images = (const float*)d_in[0];
    const float* gt     = (const float*)d_in[1];
    const float* bbw[5]; const float* bbb[5];
    for (int i = 0; i < 5; ++i) { bbw[i] = (const float*)d_in[2 + 2*i]; bbb[i] = (const float*)d_in[3 + 2*i]; }
    const float* latw[3]; const float* latb[3];
    for (int i = 0; i < 3; ++i) { latw[i] = (const float*)d_in[12 + 2*i]; latb[i] = (const float*)d_in[13 + 2*i]; }
    const float* fow[3]; const float* fob[3];
    for (int i = 0; i < 3; ++i) { fow[i] = (const float*)d_in[18 + 2*i]; fob[i] = (const float*)d_in[19 + 2*i]; }
    const float* pbw = (const float*)d_in[24]; const float* pbb = (const float*)d_in[25];
    const float* pcw = (const float*)d_in[26]; const float* pcb = (const float*)d_in[27];
    const float* ptw = (const float*)d_in[28]; const float* ptb = (const float*)d_in[29];
    const float* sbw[4]; const float* sbb[4];
    for (int i = 0; i < 4; ++i) { sbw[i] = (const float*)d_in[30 + 2*i]; sbb[i] = (const float*)d_in[31 + 2*i]; }
    const float* scw[4]; const float* scb[4];
    for (int i = 0; i < 4; ++i) { scw[i] = (const float*)d_in[38 + 2*i]; scb[i] = (const float*)d_in[39 + 2*i]; }

    // ---- workspace layout ----
    size_t off = 0;
    auto alloc = [&](size_t nbytes) -> size_t {
        size_t cur = off; off += nbytes;
        off = (off + 255) & ~(size_t)255; return cur;
    };
    const size_t F = sizeof(float), HB = sizeof(_Float16);
    const size_t oAcc = alloc(8 * F);
    const size_t oX1  = alloc((size_t)B*32*256*256 * F);   // backbone act 1
    const size_t oX2  = alloc((size_t)B*64*128*128 * F);   // backbone act 2
    const size_t oX3  = alloc((size_t)B*64*64*64   * F);
    const size_t oX4  = alloc((size_t)B*160*32*32  * F);
    const size_t oX5  = alloc((size_t)B*400*16*16  * F);
    const size_t oL3  = alloc((size_t)B*CH*64*64 * F);
    const size_t oL4  = alloc((size_t)B*CH*32*32 * F);
    const size_t oL5  = alloc((size_t)B*CH*16*16 * F);
    const size_t oL3h = alloc((size_t)B*CH*64*64 * HB);    // f16 copies of laterals
    const size_t oL4h = alloc((size_t)B*CH*32*32 * HB);
    const size_t oL5h = alloc((size_t)B*CH*16*16 * HB);
    const size_t oP16 = alloc((size_t)B*CH*64*64 * HB);    // fpn_out result (f16)
    const size_t oS0h = alloc((size_t)B*CH*64*64 * HB);    // stem ping (f16)
    const size_t oS1h = alloc((size_t)B*CH*64*64 * HB);    // stem pong (f16)
    const size_t oS32 = alloc((size_t)B*CH*64*64 * F);     // final stem act (f32)
    const size_t oPC  = alloc((size_t)B*NCLS*64*64 * F);
    const size_t oPB  = alloc((size_t)B*4*64*64 * F);
    const size_t oPT  = alloc((size_t)B*1*64*64 * F);
    const size_t oWP  = alloc((size_t)11*589824 * HB + 256); // 11 packed weight sets
    if (ws_size < off) return;

    char* ws = (char*)d_ws;
    float* acc = (float*)(ws + oAcc);
    float* X1 = (float*)(ws + oX1); float* X2 = (float*)(ws + oX2);
    float* X3 = (float*)(ws + oX3); float* X4 = (float*)(ws + oX4);
    float* X5 = (float*)(ws + oX5);
    float* L3 = (float*)(ws + oL3); float* L4 = (float*)(ws + oL4);
    float* L5 = (float*)(ws + oL5);
    _Float16* L3h = (_Float16*)(ws + oL3h);
    _Float16* L4h = (_Float16*)(ws + oL4h);
    _Float16* L5h = (_Float16*)(ws + oL5h);
    _Float16* P16 = (_Float16*)(ws + oP16);
    _Float16* S0h = (_Float16*)(ws + oS0h);
    _Float16* S1h = (_Float16*)(ws + oS1h);
    float* S32 = (float*)(ws + oS32);
    float* PC = (float*)(ws + oPC); float* PB = (float*)(ws + oPB);
    float* PT = (float*)(ws + oPT);
    _Float16* WP = (_Float16*)(ws + oWP);
    const size_t WSZ = (size_t)589824; // halves per packed conv
    _Float16* wpFO[3] = { WP + 0*WSZ, WP + 1*WSZ, WP + 2*WSZ };
    _Float16* wpSC[4] = { WP + 3*WSZ, WP + 4*WSZ, WP + 5*WSZ, WP + 6*WSZ };
    _Float16* wpSB[4] = { WP + 7*WSZ, WP + 8*WSZ, WP + 9*WSZ, WP + 10*WSZ };

    zero4_kernel<<<1, 32, 0, stream>>>(acc);

    // ---- pack all 256x256x3x3 weights to f16 fragment layout ----
    {
        const int total = CH * CH * 9;
        const int blk = (total + 255) / 256;
        for (int i = 0; i < 3; ++i) pack_w_kernel<<<blk, 256, 0, stream>>>(fow[i], wpFO[i]);
        for (int i = 0; i < 4; ++i) pack_w_kernel<<<blk, 256, 0, stream>>>(scw[i], wpSC[i]);
        for (int i = 0; i < 4; ++i) pack_w_kernel<<<blk, 256, 0, stream>>>(sbw[i], wpSB[i]);
    }

    auto conv = [&](const float* x, const float* w, const float* bias, float* y,
                    int Cin, int Hin, int Win, int Cout, int K, int stride, int relu) {
        int pad = K / 2;
        int Hout = (Hin + 2 * pad - K) / stride + 1;
        int Wout = (Win + 2 * pad - K) / stride + 1;
        int total = B * Cout * Hout * Wout;
        conv_nchw_kernel<<<(total + 255) / 256, 256, 0, stream>>>(
            x, w, bias, y, B, Cin, Hin, Win, Cout, Hout, Wout, K, stride, relu);
    };
    auto convw = [&](const _Float16* x, const _Float16* wp, const float* bias,
                     _Float16* y16, float* y32, int H, int W, int relu, int w32) {
        dim3 grid((W / 16) * 2, H, B);
        conv3x3_wmma_kernel<<<grid, 128, 0, stream>>>(x, wp, bias, y16, y32,
                                                      H, W, relu, w32);
    };

    // ---- backbone ----
    conv(images, bbw[0], bbb[0], X1,   3, 512, 512,  32, 3, 2, 1);
    conv(X1,     bbw[1], bbb[1], X2,  32, 256, 256,  64, 3, 2, 1);
    conv(X2,     bbw[2], bbb[2], X3,  64, 128, 128,  64, 3, 2, 1);
    conv(X3,     bbw[3], bbb[3], X4,  64,  64,  64, 160, 3, 2, 1);
    conv(X4,     bbw[4], bbb[4], X5, 160,  32,  32, 400, 3, 2, 1);

    // ---- FPN laterals + top-down merge + f16 conversion ----
    conv(X3, latw[0], latb[0], L3,  64, 64, 64, CH, 1, 1, 0);
    conv(X4, latw[1], latb[1], L4, 160, 32, 32, CH, 1, 1, 0);
    conv(X5, latw[2], latb[2], L5, 400, 16, 16, CH, 1, 1, 0);
    {
        int t4 = B * CH * 32 * 32;
        add_up2_kernel<<<(t4 + 255) / 256, 256, 0, stream>>>(L4, L5, B, CH, 32, 32);
        int t3 = B * CH * 64 * 64;
        add_up2_kernel<<<(t3 + 255) / 256, 256, 0, stream>>>(L3, L4, B, CH, 64, 64);
        int t5 = B * CH * 16 * 16;
        cvt_f16_kernel<<<(t3 + 255) / 256, 256, 0, stream>>>(L3, L3h, t3);
        cvt_f16_kernel<<<(t4 + 255) / 256, 256, 0, stream>>>(L4, L4h, t4);
        cvt_f16_kernel<<<(t5 + 255) / 256, 256, 0, stream>>>(L5, L5h, t5);
    }

    // ---- per-level: fpn_out -> stems (WMMA, f16) -> heads -> loss ----
    _Float16* latsH[3] = { L3h, L4h, L5h };
    const int   Hs[3] = { 64, 32, 16 };
    const float strd[3] = { 8.0f, 16.0f, 32.0f };
    const float lob[3] = { 0.0f, 64.0f, 128.0f };
    const float hib[3] = { 64.0f, 128.0f, INFINITY };

    for (int lv = 0; lv < 3; ++lv) {
        const int H = Hs[lv], W = Hs[lv];
        convw(latsH[lv], wpFO[lv], fob[lv], P16, S32, H, W, 0, 0);

        // classification stem (4x conv+relu, shared weights across levels)
        convw(P16, wpSC[0], scb[0], S1h, S32, H, W, 1, 0);
        convw(S1h, wpSC[1], scb[1], S0h, S32, H, W, 1, 0);
        convw(S0h, wpSC[2], scb[2], S1h, S32, H, W, 1, 0);
        convw(S1h, wpSC[3], scb[3], S0h, S32, H, W, 1, 1);   // + f32 for head
        conv(S32, pcw, pcb, PC, CH, H, W, NCLS, 3, 1, 0);

        // box/centerness stem
        convw(P16, wpSB[0], sbb[0], S1h, S32, H, W, 1, 0);
        convw(S1h, wpSB[1], sbb[1], S0h, S32, H, W, 1, 0);
        convw(S0h, wpSB[2], sbb[2], S1h, S32, H, W, 1, 0);
        convw(S1h, wpSB[3], sbb[3], S0h, S32, H, W, 1, 1);   // + f32 for heads
        conv(S32, pbw, pbb, PB, CH, H, W, 4, 3, 1, 0);
        conv(S32, ptw, ptb, PT, CH, H, W, 1, 3, 1, 0);

        dim3 lgrid((H * W + 255) / 256, B);
        fcos_loss_kernel<<<lgrid, 256, 0, stream>>>(
            PC, PB, PT, gt, H, W, strd[lv], lob[lv], hib[lv], acc);
    }

    finalize_kernel<<<1, 32, 0, stream>>>(acc, (float*)d_out);
}
